// UniTR_67826123538975
// MI455X (gfx1250) — matrix operand order repaired and checked by reading, not compiled, not run on hardware
//
#include <hip/hip_runtime.h>
#include <hip/hip_bf16.h>
#include <stdint.h>

// ---------------------------------------------------------------------------
// UniTR 2-layer set-transformer for MI455X (gfx1250, wave32, WMMA).
// GEMMs use v_wmma_f32_16x16x32_f16; activation tiles are fed to LDS via the
// CDNA5 async global->LDS path (ASYNCcnt) when the toolchain exposes it.
// ---------------------------------------------------------------------------

typedef __attribute__((ext_vector_type(16))) _Float16 v16h;
typedef __attribute__((ext_vector_type(8)))  _Float16 v8h;
typedef __attribute__((ext_vector_type(8)))  float    v8f;
typedef __attribute__((ext_vector_type(4)))  int      v4i;

#define DEV static __device__ __forceinline__

constexpr int NVOX  = 65536;   // N voxels
constexpr int DM    = 256;     // d_model
constexpr int NH    = 8;       // heads
constexpr int HD    = 32;      // head dim
constexpr int DFF   = 2048;    // ffn dim
constexpr int SS    = 128;     // set size
constexpr int SN    = 512;     // set num
constexpr int LROWS = SN * SS; // 65536 gathered rows

// ---- async global->LDS (GLOBAL_LOAD_ASYNC_TO_LDS_B128 / s_wait_asynccnt) ----
#if __has_builtin(__builtin_amdgcn_global_load_async_to_lds_b128) && \
    __has_builtin(__builtin_amdgcn_s_wait_asynccnt)
#define HAVE_ASYNC_LDS 1
#else
#define HAVE_ASYNC_LDS 0
#endif

typedef __attribute__((address_space(1))) v4i gas_v4i;  // global int4
typedef __attribute__((address_space(3))) v4i las_v4i;  // LDS int4

DEV void async_copy16(const void* gsrc, void* ldst) {
#if HAVE_ASYNC_LDS
  __builtin_amdgcn_global_load_async_to_lds_b128(
      (gas_v4i*)gsrc, (las_v4i*)ldst, 0, 0);
#else
  *(v8h*)ldst = *(const v8h*)gsrc;
#endif
}
DEV void async_wait() {
#if HAVE_ASYNC_LDS
  __builtin_amdgcn_s_wait_asynccnt(0);
#endif
}

// ---- WMMA fragment loaders (layouts per CDNA5 ISA §7.12.2) ----
// A (16x32 f16): lanes 0-15 hold row M, elems 0..7 = K0..7, 8..15 = K16..23;
//                lanes 16-31 hold K8..15 / K24..31.  LDS A is row-major [M][>=32].
DEV v16h load_frag_a(const _Float16* A, int M0, int stride) {
  const int lane  = threadIdx.x & 31;
  const int l     = lane & 15;
  const int hiOff = (lane >> 4) << 3;                 // 0 or 8 halves
  const _Float16* p = A + (size_t)(M0 + l) * stride + hiOff;
  union { v16h v; v8h h[2]; } u;
  u.h[0] = *reinterpret_cast<const v8h*>(p);
  u.h[1] = *reinterpret_cast<const v8h*>(p + 16);
  return u.v;
}
// B (32x16 f16): lanes 0-15 hold col N, elems 0..7 = K0..7, 8..15 = K16..23;
//                lanes 16-31 hold K8..15 / K24..31.
// LDS B is stored as [N][K] (i.e. W's natural row-major layout).
DEV v16h load_frag_b(const _Float16* B, int N0, int stride) {
  const int lane  = threadIdx.x & 31;
  const int l     = lane & 15;
  const int hiOff = (lane >> 4) << 3;                 // 0 or 8 halves
  const _Float16* p = B + (size_t)(N0 + l) * stride + hiOff;
  union { v16h v; v8h h[2]; } u;
  u.h[0] = *reinterpret_cast<const v8h*>(p);
  u.h[1] = *reinterpret_cast<const v8h*>(p + 16);
  return u.v;
}
DEV v8f wmma16(v16h a, v16h b, v8f c) {
  return __builtin_amdgcn_wmma_f32_16x16x32_f16(false, a, false, b,
                                                (short)0, c, false, false);
}
// C/D f32 16x16: VGPR r -> lanes 0-15: (M0+r, N0+lane); lanes 16-31: (M0+8+r, N0+lane-16)

// ---------------------------------------------------------------------------
// 1) gather: feat16 = out[inds], qk16 = out[inds] + pos[inds]   (f32 -> f16)
// ---------------------------------------------------------------------------
__global__ __launch_bounds__(256) void gather_kernel(
    const float* __restrict__ out_in, const float* __restrict__ pos,
    const int* __restrict__ inds,
    _Float16* __restrict__ qk16, _Float16* __restrict__ feat16) {
  int g = blockIdx.x * blockDim.x + threadIdx.x;      // one thread per 8 elems
  int r = g >> 5;
  int d = (g & 31) * 8;
  int v = inds[r];
  const float* sp = out_in + (size_t)v * DM + d;
  const float* pp = pos    + (size_t)v * DM + d;
  _Float16* q = qk16   + (size_t)r * DM + d;
  _Float16* f = feat16 + (size_t)r * DM + d;
#pragma unroll
  for (int j = 0; j < 8; j++) {
    float s = sp[j];
    f[j] = (_Float16)s;
    q[j] = (_Float16)(s + pp[j]);
  }
}

// ---------------------------------------------------------------------------
// perm: first occurrence of each voxel in flattened set order
// ---------------------------------------------------------------------------
__global__ __launch_bounds__(256) void perm_init_kernel(int* __restrict__ perm) {
  perm[blockIdx.x * blockDim.x + threadIdx.x] = LROWS;
}
__global__ __launch_bounds__(256) void perm_min_kernel(
    const int* __restrict__ inds, int* __restrict__ perm) {
  int l = blockIdx.x * blockDim.x + threadIdx.x;
  atomicMin(&perm[inds[l]], l);
}

// ---------------------------------------------------------------------------
// 2) QKV GEMM: [L x 256] @ W^T[256 x 768] -> Q/K/V in head-major f16.
//    Block tile 128(M) x 64(N), 8 waves each doing a 32x32 subtile.
//    Cols <512 (Q,K) read qk16; cols >=512 (V) read feat16.
// ---------------------------------------------------------------------------
__global__ __launch_bounds__(256) void qkv_gemm_kernel(
    const _Float16* __restrict__ qk16, const _Float16* __restrict__ feat16,
    const float* __restrict__ W, const float* __restrict__ bias,
    _Float16* __restrict__ Qh, _Float16* __restrict__ Kh, _Float16* __restrict__ Vh) {
  __shared__ _Float16 Alds[128 * 32];
  __shared__ _Float16 Blds[64 * 32];
  const int tid = threadIdx.x, lane = tid & 31, wave = tid >> 5;
  const int wm = wave & 3, wn = wave >> 2;
  const int rB = blockIdx.y * 128, nB = blockIdx.x * 64;
  const _Float16* A = (nB < 2 * DM) ? qk16 : feat16;
  v8f acc[2][2] = {};
  const int arow = tid >> 1, aoff = (tid & 1) * 16;
  const int brow = tid >> 2, boff = (tid & 3) * 8;
  for (int kc = 0; kc < DM; kc += 32) {
    const _Float16* ga = A + (size_t)(rB + arow) * DM + kc + aoff;
    _Float16* la = Alds + arow * 32 + aoff;
    async_copy16(ga, la);                       // ASYNCcnt-tracked tile fill
    async_copy16(ga + 8, la + 8);
    const float* gb = W + (size_t)(nB + brow) * DM + kc + boff;
    if (kc + 32 < DM) __builtin_prefetch(gb + 32, 0, 1);  // global_prefetch_b8
#pragma unroll
    for (int j = 0; j < 8; j++) Blds[brow * 32 + boff + j] = (_Float16)gb[j];
    async_wait();
    __syncthreads();
#pragma unroll
    for (int tm = 0; tm < 2; tm++) {
      v16h a = load_frag_a(Alds, wm * 32 + tm * 16, 32);
#pragma unroll
      for (int tn = 0; tn < 2; tn++) {
        v16h b = load_frag_b(Blds, wn * 32 + tn * 16, 32);
        acc[tm][tn] = wmma16(a, b, acc[tm][tn]);
      }
    }
    __syncthreads();
  }
  const int l = lane & 15, hi8 = (lane >> 4) * 8;
#pragma unroll
  for (int tm = 0; tm < 2; tm++)
#pragma unroll
    for (int tn = 0; tn < 2; tn++) {
      int gc = nB + wn * 32 + tn * 16 + l;
      float bi = bias[gc];
#pragma unroll
      for (int ri = 0; ri < 8; ri++) {
        int gr = rB + wm * 32 + tm * 16 + hi8 + ri;
        float val = acc[tm][tn][ri] + bi;
        int set = gr >> 7, q = gr & 127;
        _Float16 hv = (_Float16)val;
        if (gc < DM) {
          Qh[(((size_t)set * NH + (gc >> 5)) * SS + q) * HD + (gc & 31)] = hv;
        } else if (gc < 2 * DM) {
          int c = gc - DM;
          Kh[(((size_t)set * NH + (c >> 5)) * SS + q) * HD + (c & 31)] = hv;
        } else {
          int c = gc - 2 * DM;
          Vh[(((size_t)set * NH + (c >> 5)) * SS + q) * HD + (c & 31)] = hv;
        }
      }
    }
}

// ---------------------------------------------------------------------------
// 3) attention: one block per (set, head). 120 KB dynamic LDS.
//    scores = Q K^T / sqrt(hd) (1 WMMA per tile, K=32), masked softmax (f32),
//    o = attn @ V (K=128 -> 4 WMMA steps).
// ---------------------------------------------------------------------------
__global__ __launch_bounds__(256) void attn_kernel(
    const _Float16* __restrict__ Qh, const _Float16* __restrict__ Kh,
    const _Float16* __restrict__ Vh, const unsigned char* __restrict__ mask,
    _Float16* __restrict__ o16) {
  extern __shared__ char smem[];
  float*    sc   = (float*)smem;                                  // 128*128 f32
  _Float16* Ql   = (_Float16*)(smem + 65536);                     // 128*32
  _Float16* Kl   = (_Float16*)(smem + 65536 + 8192);              // 128*32
  _Float16* Vt   = (_Float16*)(smem + 65536 + 16384);             // 32*128 (V^T)
  _Float16* at   = (_Float16*)(smem + 65536 + 24576);             // 128*128 f16
  float*    madd = (float*)(smem + 65536 + 24576 + 32768);        // 128 f32
  const int h = blockIdx.x, s = blockIdx.y;
  const int tid = threadIdx.x, lane = tid & 31, wave = tid >> 5;
  const int l = lane & 15, hi8 = (lane >> 4) * 8;
  const size_t base = (((size_t)s * NH) + h) * SS * HD;
  {
    int row = tid >> 1, off = (tid & 1) * 16;
    const _Float16* gq = Qh + base + (size_t)row * HD + off;
    const _Float16* gk = Kh + base + (size_t)row * HD + off;
    _Float16* sq = Ql + row * HD + off;
    _Float16* sk = Kl + row * HD + off;
    async_copy16(gq, sq); async_copy16(gq + 8, sq + 8);
    async_copy16(gk, sk); async_copy16(gk + 8, sk + 8);
  }
#pragma unroll
  for (int j = 0; j < 16; j++) {                                   // V transpose
    int e = tid * 16 + j; int q = e >> 5, d = e & 31;
    Vt[d * SS + q] = Vh[base + e];
  }
  if (tid < SS) madd[tid] = mask[(size_t)s * SS + tid] ? 1.f : 0.f;
  async_wait();
  __syncthreads();
  const float scale = 0.17677669529663687f;                        // 1/sqrt(32)
  {
    int M0 = wave * 16;
    v16h a = load_frag_a(Ql, M0, HD);
    for (int tn = 0; tn < 8; tn++) {
      v16h b = load_frag_b(Kl, tn * 16, HD);                       // B = K^T
      v8f c = {};
      c = wmma16(a, b, c);
#pragma unroll
      for (int ri = 0; ri < 8; ri++) {
        int row = M0 + hi8 + ri, col = tn * 16 + l;
        sc[row * SS + col] = (madd[col] != 0.f) ? -1e9f : c[ri] * scale;
      }
    }
  }
  __syncthreads();
  if (tid < SS) {                                                  // softmax (f32)
    float m = -3.4e38f;
    for (int c2 = 0; c2 < SS; c2++) m = fmaxf(m, sc[tid * SS + c2]);
    float sum = 0.f;
    for (int c2 = 0; c2 < SS; c2++) {
      float e = __expf(sc[tid * SS + c2] - m);
      sc[tid * SS + c2] = e; sum += e;
    }
    float inv = 1.f / sum;
    for (int c2 = 0; c2 < SS; c2++) at[tid * SS + c2] = (_Float16)(sc[tid * SS + c2] * inv);
  }
  __syncthreads();
  {
    int M0 = wave * 16;
    v8f acc[2] = {};
    for (int kc = 0; kc < SS; kc += 32) {
      v16h a = load_frag_a(at + kc, M0, SS);
#pragma unroll
      for (int tn = 0; tn < 2; tn++) {
        v16h b = load_frag_b(Vt + kc, tn * 16, SS);
        acc[tn] = wmma16(a, b, acc[tn]);
      }
    }
#pragma unroll
    for (int tn = 0; tn < 2; tn++)
#pragma unroll
      for (int ri = 0; ri < 8; ri++) {
        int q = M0 + hi8 + ri, d = tn * 16 + l;
        o16[((size_t)(s * SS + q)) * DM + h * HD + d] = (_Float16)acc[tn][ri];
      }
  }
}

// ---------------------------------------------------------------------------
// 4) generic projection GEMM: C[L x ncols](f32) = A[L x 256](f16) @ W^T + bias
// ---------------------------------------------------------------------------
__global__ __launch_bounds__(256) void proj_gemm_kernel(
    const _Float16* __restrict__ A, const float* __restrict__ W,
    const float* __restrict__ bias, float* __restrict__ C, int ncols) {
  __shared__ _Float16 Alds[128 * 32];
  __shared__ _Float16 Blds[64 * 32];
  const int tid = threadIdx.x, lane = tid & 31, wave = tid >> 5;
  const int wm = wave & 3, wn = wave >> 2;
  const int rB = blockIdx.y * 128, nB = blockIdx.x * 64;
  v8f acc[2][2] = {};
  const int arow = tid >> 1, aoff = (tid & 1) * 16;
  const int brow = tid >> 2, boff = (tid & 3) * 8;
  for (int kc = 0; kc < DM; kc += 32) {
    const _Float16* ga = A + (size_t)(rB + arow) * DM + kc + aoff;
    _Float16* la = Alds + arow * 32 + aoff;
    async_copy16(ga, la);
    async_copy16(ga + 8, la + 8);
    const float* gb = W + (size_t)(nB + brow) * DM + kc + boff;
    if (kc + 32 < DM) __builtin_prefetch(gb + 32, 0, 1);
#pragma unroll
    for (int j = 0; j < 8; j++) Blds[brow * 32 + boff + j] = (_Float16)gb[j];
    async_wait();
    __syncthreads();
#pragma unroll
    for (int tm = 0; tm < 2; tm++) {
      v16h a = load_frag_a(Alds, wm * 32 + tm * 16, 32);
#pragma unroll
      for (int tn = 0; tn < 2; tn++) {
        v16h b = load_frag_b(Blds, wn * 32 + tn * 16, 32);
        acc[tm][tn] = wmma16(a, b, acc[tm][tn]);
      }
    }
    __syncthreads();
  }
  const int l = lane & 15, hi8 = (lane >> 4) * 8;
#pragma unroll
  for (int tm = 0; tm < 2; tm++)
#pragma unroll
    for (int tn = 0; tn < 2; tn++) {
      int gc = nB + wn * 32 + tn * 16 + l;
      float bi = bias[gc];
#pragma unroll
      for (int ri = 0; ri < 8; ri++) {
        int gr = rB + wm * 32 + tm * 16 + hi8 + ri;
        C[(size_t)gr * ncols + gc] = acc[tm][tn][ri] + bi;
      }
    }
}

// ---------------------------------------------------------------------------
// pointwise LayerNorm helpers
// ---------------------------------------------------------------------------
DEV void reduce2(float& s, float& s2, float* sA, float* sB) {
  int d = threadIdx.x;
  __syncthreads();
  sA[d] = s; sB[d] = s2; __syncthreads();
  for (int k = 128; k > 0; k >>= 1) {
    if (d < k) { sA[d] += sA[d + k]; sB[d] += sB[d + k]; }
    __syncthreads();
  }
  s = sA[0]; s2 = sB[0];
}

// 5) x = LN1(out + a[perm[v]]) -> x32 (f32) and x16 (f16)
__global__ __launch_bounds__(256) void ln1_kernel(
    const float* __restrict__ out_in, const float* __restrict__ aG,
    const int* __restrict__ perm, const float* __restrict__ g,
    const float* __restrict__ b, float* __restrict__ x32,
    _Float16* __restrict__ x16) {
  __shared__ float sA[256], sB[256];
  int v = blockIdx.x, d = threadIdx.x;
  int pr = perm[v];
  float val = out_in[(size_t)v * DM + d] + aG[(size_t)pr * DM + d];
  float s = val, s2 = val * val;
  reduce2(s, s2, sA, sB);
  float m = s / DM, var = s2 / DM - m * m;
  float y = (val - m) * rsqrtf(var + 1e-5f) * g[d] + b[d];
  x32[(size_t)v * DM + d] = y;
  x16[(size_t)v * DM + d] = (_Float16)y;
}

// ---------------------------------------------------------------------------
// 6) fused FFN: ff = relu(x @ W1^T + b1) @ W2^T + b2, streamed over DFF in
//    64-wide chunks; x tile (128x256 f16) resident in LDS; f32 WMMA acc.
// ---------------------------------------------------------------------------
__global__ __launch_bounds__(256) void ffn_kernel(
    const _Float16* __restrict__ x16, const float* __restrict__ W1,
    const float* __restrict__ B1v, const float* __restrict__ W2,
    const float* __restrict__ B2v, float* __restrict__ ffG) {
  extern __shared__ char smem[];
  _Float16* xt  = (_Float16*)smem;                         // 128*256 = 64KB
  _Float16* b1l = (_Float16*)(smem + 65536);               // 64*32   =  4KB
  _Float16* hl  = (_Float16*)(smem + 65536 + 4096);        // 128*64  = 16KB
  _Float16* b2l = (_Float16*)(smem + 65536 + 4096 + 16384);// 256*32  = 16KB
  const int tid = threadIdx.x, lane = tid & 31, wave = tid >> 5;
  const int wm = wave & 3, wn = wave >> 2;
  const int l = lane & 15, hi8 = (lane >> 4) * 8;
  const int rB = blockIdx.x * 128;
  {
    int row = tid >> 1, off = (tid & 1) * 128;
    const _Float16* g = x16 + (size_t)(rB + row) * DM + off;
    _Float16* sdst = xt + row * DM + off;
#pragma unroll
    for (int j = 0; j < 16; j++) async_copy16(g + j * 8, sdst + j * 8);
    async_wait();
  }
  v8f acc2[2][8] = {};
  for (int dc = 0; dc < DFF; dc += 64) {
    __syncthreads();
    v8f acc1[2][2] = {};
    for (int kc = 0; kc < DM; kc += 32) {
      { int n = tid >> 2, koff = (tid & 3) * 8;
        const float* g = W1 + (size_t)(dc + n) * DM + kc + koff;
        if (kc + 32 < DM) __builtin_prefetch(g + 32, 0, 1);
#pragma unroll
        for (int j = 0; j < 8; j++) b1l[n * 32 + koff + j] = (_Float16)g[j]; }
      __syncthreads();
#pragma unroll
      for (int tm = 0; tm < 2; tm++) {
        v16h a = load_frag_a(xt + kc, wm * 32 + tm * 16, DM);
#pragma unroll
        for (int tn = 0; tn < 2; tn++) {
          v16h b = load_frag_b(b1l, wn * 32 + tn * 16, 32);
          acc1[tm][tn] = wmma16(a, b, acc1[tm][tn]);
        }
      }
      __syncthreads();
    }
#pragma unroll
    for (int tm = 0; tm < 2; tm++)
#pragma unroll
      for (int tn = 0; tn < 2; tn++) {
        int col = wn * 32 + tn * 16 + l;
        float bi = B1v[dc + col];
#pragma unroll
        for (int ri = 0; ri < 8; ri++) {
          int row = wm * 32 + tm * 16 + hi8 + ri;
          float hv = acc1[tm][tn][ri] + bi;
          hl[row * 64 + col] = (_Float16)(hv > 0.f ? hv : 0.f);
        }
      }
    __syncthreads();
    for (int ks = 0; ks < 2; ks++) {
      { int n = tid;
        const float* g = W2 + (size_t)n * DFF + dc + ks * 32;
        __builtin_prefetch(g + 32, 0, 1);
#pragma unroll
        for (int j = 0; j < 32; j++) b2l[n * 32 + j] = (_Float16)g[j]; }
      __syncthreads();
#pragma unroll
      for (int tm = 0; tm < 2; tm++) {
        v16h a = load_frag_a(hl + ks * 32, wm * 32 + tm * 16, 64);
#pragma unroll
        for (int tn = 0; tn < 8; tn++) {
          v16h b = load_frag_b(b2l, wn * 128 + tn * 16, 32);
          acc2[tm][tn] = wmma16(a, b, acc2[tm][tn]);
        }
      }
      __syncthreads();
    }
  }
#pragma unroll
  for (int tm = 0; tm < 2; tm++)
#pragma unroll
    for (int tn = 0; tn < 8; tn++) {
      int gc = wn * 128 + tn * 16 + l;
      float bi = B2v[gc];
#pragma unroll
      for (int ri = 0; ri < 8; ri++) {
        int gr = rB + wm * 32 + tm * 16 + hi8 + ri;
        ffG[(size_t)gr * DM + gc] = acc2[tm][tn][ri] + bi;
      }
    }
}

// 7) out = LN( LN2(x + ff) + identity )
__global__ __launch_bounds__(256) void ln2f_kernel(
    const float* __restrict__ x32, const float* __restrict__ ffG,
    const float* __restrict__ identity, const float* __restrict__ g2,
    const float* __restrict__ b2, const float* __restrict__ gf,
    const float* __restrict__ bf, float* __restrict__ outDst) {
  __shared__ float sA[256], sB[256];
  int v = blockIdx.x, d = threadIdx.x;
  float t = x32[(size_t)v * DM + d] + ffG[(size_t)v * DM + d];
  float s = t, s2 = t * t;
  reduce2(s, s2, sA, sB);
  float m = s / DM, var = s2 / DM - m * m;
  float z = (t - m) * rsqrtf(var + 1e-5f) * g2[d] + b2[d];
  float o = z + identity[(size_t)v * DM + d];
  s = o; s2 = o * o;
  reduce2(s, s2, sA, sB);
  m = s / DM; var = s2 / DM - m * m;
  outDst[(size_t)v * DM + d] = (o - m) * rsqrtf(var + 1e-5f) * gf[d] + bf[d];
}

// ---------------------------------------------------------------------------
extern "C" void kernel_launch(void* const* d_in, const int* in_sizes, int n_in,
                              void* d_out, int out_size, void* d_ws, size_t ws_size,
                              hipStream_t stream) {
  (void)in_sizes; (void)n_in; (void)out_size; (void)ws_size;
  const float* src   = (const float*)d_in[0];
  const float* pos   = (const float*)d_in[1];
  const int*   inds  = (const int*)d_in[2];
  const unsigned char* masks = (const unsigned char*)d_in[3];   // numpy bool
  const float* w_qkv = (const float*)d_in[4];
  const float* b_qkv = (const float*)d_in[5];
  const float* w_out = (const float*)d_in[6];
  const float* b_out = (const float*)d_in[7];
  const float* w_ff1 = (const float*)d_in[8];
  const float* b_ff1 = (const float*)d_in[9];
  const float* w_ff2 = (const float*)d_in[10];
  const float* b_ff2 = (const float*)d_in[11];
  const float* g_n1  = (const float*)d_in[12];
  const float* b_n1  = (const float*)d_in[13];
  const float* g_n2  = (const float*)d_in[14];
  const float* b_n2  = (const float*)d_in[15];
  const float* g_ln  = (const float*)d_in[16];
  const float* b_ln  = (const float*)d_in[17];
  float* outF = (float*)d_out;

  char* ws = (char*)d_ws; size_t off = 0;
  auto carve = [&](size_t bytes) -> char* {
    char* p = ws + off; off += (bytes + 255) & ~(size_t)255; return p;
  };
  _Float16* qk16   = (_Float16*)carve((size_t)LROWS * DM * 2);
  _Float16* feat16 = (_Float16*)carve((size_t)LROWS * DM * 2);
  _Float16* Qh     = (_Float16*)carve((size_t)LROWS * DM * 2);
  _Float16* Kh     = (_Float16*)carve((size_t)LROWS * DM * 2);
  _Float16* Vh     = (_Float16*)carve((size_t)LROWS * DM * 2);
  float*    aG     = (float*)carve((size_t)LROWS * DM * 4);
  float*    x32    = (float*)carve((size_t)NVOX * DM * 4);
  int*      perm   = (int*)carve((size_t)NVOX * 4);
  _Float16* o16 = qk16;    // lifetime-safe aliases
  _Float16* x16 = feat16;
  float*    ffG = aG;

  const size_t ATTN_LDS = 65536 + 8192 + 8192 + 8192 + 32768 + 512; // 123,392 B
  const size_t FFN_LDS  = 65536 + 4096 + 16384 + 16384;             // 102,400 B
  (void)hipFuncSetAttribute((const void*)attn_kernel,
                      hipFuncAttributeMaxDynamicSharedMemorySize, (int)ATTN_LDS);
  (void)hipFuncSetAttribute((const void*)ffn_kernel,
                      hipFuncAttributeMaxDynamicSharedMemorySize, (int)FFN_LDS);

  for (int i = 0; i < 2; i++) {
    const float* out_in = (i == 0) ? src : outF;
    const int* inds_i = inds + (size_t)i * 2 * SN * SS;           // set_id = 0
    const unsigned char* mask_i = masks + (size_t)i * 2 * SN * SS;
    const float* pos_i  = pos + (size_t)i * NVOX * DM;
    const float* wqkv_i = w_qkv + (size_t)i * 3 * DM * DM;
    const float* bqkv_i = b_qkv + (size_t)i * 3 * DM;
    const float* wout_i = w_out + (size_t)i * DM * DM;
    const float* bout_i = b_out + (size_t)i * DM;
    const float* w1_i   = w_ff1 + (size_t)i * DFF * DM;
    const float* b1_i   = b_ff1 + (size_t)i * DFF;
    const float* w2_i   = w_ff2 + (size_t)i * DM * DFF;
    const float* b2_i   = b_ff2 + (size_t)i * DM;
    const float* gn1_i = g_n1 + (size_t)i * DM; const float* bn1_i = b_n1 + (size_t)i * DM;
    const float* gn2_i = g_n2 + (size_t)i * DM; const float* bn2_i = b_n2 + (size_t)i * DM;
    const float* gln_i = g_ln + (size_t)i * DM; const float* bln_i = b_ln + (size_t)i * DM;

    gather_kernel<<<(LROWS * (DM / 8)) / 256, 256, 0, stream>>>(
        out_in, pos_i, inds_i, qk16, feat16);
    perm_init_kernel<<<NVOX / 256, 256, 0, stream>>>(perm);
    perm_min_kernel<<<LROWS / 256, 256, 0, stream>>>(inds_i, perm);
    qkv_gemm_kernel<<<dim3(3 * DM / 64, LROWS / 128), 256, 0, stream>>>(
        qk16, feat16, wqkv_i, bqkv_i, Qh, Kh, Vh);
    attn_kernel<<<dim3(NH, SN), 256, ATTN_LDS, stream>>>(Qh, Kh, Vh, mask_i, o16);
    proj_gemm_kernel<<<dim3(DM / 64, LROWS / 128), 256, 0, stream>>>(
        o16, wout_i, bout_i, aG, DM);
    ln1_kernel<<<NVOX, 256, 0, stream>>>(out_in, aG, perm, gn1_i, bn1_i, x32, x16);
    ffn_kernel<<<NVOX / 128, 256, FFN_LDS, stream>>>(x16, w1_i, b1_i, w2_i, b2_i, ffG);
    ln2f_kernel<<<NVOX, 256, 0, stream>>>(x32, ffG, out_in, gn2_i, bn2_i,
                                          gln_i, bln_i, outF);
  }
}
